// FourierNGP_44607530336772
// MI455X (gfx1250) — matrix operand order repaired and checked by reading, not compiled, not run on hardware
//
#include <hip/hip_runtime.h>

// ---------------------------------------------------------------------------
// FourierNGP SIREN MLP [74,128,128,128,3] on gfx1250, v_wmma_f32_16x16x32_f16.
//
// Round-2 changes vs round-1:
//  * M=32 points per wave (2 A row-tiles) -> each B fragment (1KB) feeds 2
//    WMMAs, halving weight-fetch bandwidth (the binding operand stream).
//  * Single in-place LDS activation buffer per wave (A fragments fully
//    register-resident before stores; LDS ops are in-order within a wave).
//    LDS: 4 waves * 32 rows * 136 f16 * 2B = 34.8 KB.
// ---------------------------------------------------------------------------

typedef _Float16 v16h __attribute__((ext_vector_type(16)));
typedef _Float16 v8h  __attribute__((ext_vector_type(8)));
typedef _Float16 v2h  __attribute__((ext_vector_type(2)));
typedef float    v8f  __attribute__((ext_vector_type(8)));

#define LDS_STRIDE 136        // 128 + 8 f16 pad: 272B rows -> conflict-minimal b128 loads
#define WAVES_PER_BLOCK 4
#define NUM_FRAGS 92          // 24 (L0) + 32 (L1) + 32 (L2) + 4 (L3)
#define TILE_M 32             // points per wave
#define MT 2                  // 16-row WMMA tiles per wave

__device__ __forceinline__ void lds_fence() {
  // intra-wave LDS producer->consumer ordering (wave owns its buffer; no barrier)
  asm volatile("s_wait_dscnt 0" ::: "memory");
}

// ---------------------------------------------------------------------------
// Pre-kernel: build WMMA B-fragments.
// Fragment f, lane l holds 16 f16: column n = nt*16 + (l&15),
// K = kt*32 + (l>>4)*16 + j  (j = 0..15), zero-padded outside [Kact, Nact].
// Storage: wf[(f*32 + l)*16 + j]  -> each fragment is a coalesced 1KB block.
// ---------------------------------------------------------------------------
__global__ void build_wfrags(const float* __restrict__ W0, const float* __restrict__ W1,
                             const float* __restrict__ W2, const float* __restrict__ W3,
                             _Float16* __restrict__ wf) {
  int gid = blockIdx.x * blockDim.x + threadIdx.x;
  if (gid >= NUM_FRAGS * 32) return;
  int frag = gid >> 5;
  int lane = gid & 31;

  const float* W; int kt, nt, Kact, Nact, Nstride;
  if (frag < 24)      { W = W0; int f = frag;      kt = f >> 3; nt = f & 7; Kact = 74;  Nact = 128; Nstride = 128; }
  else if (frag < 56) { W = W1; int f = frag - 24; kt = f >> 3; nt = f & 7; Kact = 128; Nact = 128; Nstride = 128; }
  else if (frag < 88) { W = W2; int f = frag - 56; kt = f >> 3; nt = f & 7; Kact = 128; Nact = 128; Nstride = 128; }
  else                { W = W3; int f = frag - 88; kt = f;      nt = 0;     Kact = 128; Nact = 3;   Nstride = 3;   }

  int n     = nt * 16 + (lane & 15);
  int kbase = kt * 32 + (lane >> 4) * 16;
  _Float16* dst = wf + ((size_t)frag * 32 + lane) * 16;
#pragma unroll
  for (int j = 0; j < 16; ++j) {
    int k = kbase + j;
    float v = (k < Kact && n < Nact) ? W[(size_t)k * Nstride + n] : 0.0f;
    dst[j] = (_Float16)v;
  }
}

// ---------------------------------------------------------------------------
// One layer over a 32-row tile, in-place in `buf`:
//   all A fragments register-loaded first, then per n-tile:
//   C=bias; accumulate KT WMMAs per M-tile; sin/clip; store.
// ---------------------------------------------------------------------------
template <int KT, int NT, bool FINAL>
__device__ __forceinline__ void layer(_Float16* __restrict__ buf,
                                      const _Float16* __restrict__ wf, int fragBase,
                                      const float* __restrict__ bias,
                                      int lane, float* __restrict__ out, long pbase) {
  const int m  = lane & 15;   // row-in-tile / column index
  const int hi = lane >> 4;   // half-wave selector

  // A fragments (16-bit A 16x32 layout): element j = K hi*16 + j (contiguous, matches B)
  // Loaded for BOTH M-tiles before any store -> in-place buffer is safe
  // (LDS ops are in-order within a wave).
  v16h a[MT][KT];
#pragma unroll
  for (int mt = 0; mt < MT; ++mt) {
#pragma unroll
    for (int kt = 0; kt < KT; ++kt) {
      const _Float16* rp = buf + (mt * 16 + m) * LDS_STRIDE + kt * 32 + hi * 16;
      v8h lo = *(const v8h*)(rp);
      v8h hh = *(const v8h*)(rp + 8);
      a[mt][kt] = __builtin_shufflevector(lo, hh, 0,1,2,3,4,5,6,7,8,9,10,11,12,13,14,15);
    }
  }

#pragma unroll
  for (int nt = 0; nt < NT; ++nt) {
    const int n = nt * 16 + m;
    float bv = FINAL ? (n < 3 ? bias[n] : 0.0f) : bias[n];
    v8f c[MT];
#pragma unroll
    for (int mt = 0; mt < MT; ++mt) c[mt] = (v8f){bv, bv, bv, bv, bv, bv, bv, bv};
#pragma unroll
    for (int kt = 0; kt < KT; ++kt) {
      // one 1KB B fragment feeds MT WMMAs (halved weight-fetch bandwidth)
      const v16h b = *(const v16h*)(wf + ((size_t)(fragBase + kt * NT + nt) * 32 + lane) * 16);
#pragma unroll
      for (int mt = 0; mt < MT; ++mt) {
        c[mt] = __builtin_amdgcn_wmma_f32_16x16x32_f16(false, a[mt][kt], false, b,
                                                       (short)0, c[mt], false, false);
      }
    }
    if (!FINAL) {
      // D layout: VGPR r -> row (mt*16 + r + hi*8), column n
#pragma unroll
      for (int mt = 0; mt < MT; ++mt) {
#pragma unroll
        for (int r = 0; r < 8; ++r) {
          float y = __sinf(30.0f * c[mt][r]);
          buf[(mt * 16 + r + hi * 8) * LDS_STRIDE + n] = (_Float16)y;
        }
      }
    } else {
      if (n < 3) {
#pragma unroll
        for (int mt = 0; mt < MT; ++mt) {
#pragma unroll
          for (int r = 0; r < 8; ++r) {
            float y = fminf(fmaxf(c[mt][r], -1.0f), 1.0f);
            out[(pbase + mt * 16 + r + hi * 8) * 3 + n] = y;
          }
        }
      }
    }
  }
}

// ---------------------------------------------------------------------------
// Main kernel: 4 waves/block, one 32-point tile per wave.
// ---------------------------------------------------------------------------
__global__ __launch_bounds__(WAVES_PER_BLOCK * 32)
void fourier_ngp_mlp(const float* __restrict__ pos, const float* __restrict__ grid,
                     const float* __restrict__ b0, const float* __restrict__ b1,
                     const float* __restrict__ b2, const float* __restrict__ b3,
                     const _Float16* __restrict__ wf, float* __restrict__ out,
                     int nPoints) {
  __shared__ __attribute__((aligned(16))) _Float16 smem[WAVES_PER_BLOCK][TILE_M * LDS_STRIDE];

  const int lane = threadIdx.x & 31;
  const int wave = threadIdx.x >> 5;
  const int m    = lane & 15;
  const int hi   = lane >> 4;

  const long tile  = (long)blockIdx.x * WAVES_PER_BLOCK + wave;
  const long pbase = tile * TILE_M;
  if (pbase >= nPoints) return;   // wave-uniform

  _Float16* buf = smem[wave];

  // ---- build input features [32 x 96] into buf (f16) ----
  // cols 0..31 grid, 32..33 pos, 34..43 sin(x0*f), 44..53 cos(x0*f),
  // 54..63 sin(x1*f), 64..73 cos(x1*f), 74..95 zero pad
#pragma unroll
  for (int t = 0; t < MT; ++t) {
    const int  rm = t * 16 + m;
    const long p  = pbase + rm;
    const float x0 = pos[p * 2 + 0];
    const float x1 = pos[p * 2 + 1];
    _Float16* row = buf + rm * LDS_STRIDE;

    if (hi == 0) {
      const float4* g4 = (const float4*)(grid + p * 32);
#pragma unroll
      for (int j = 0; j < 8; ++j) {
        float4 g = g4[j];
        v2h p0 = {(_Float16)g.x, (_Float16)g.y};
        v2h p1 = {(_Float16)g.z, (_Float16)g.w};
        *(v2h*)(row + j * 4)     = p0;
        *(v2h*)(row + j * 4 + 2) = p1;
      }
      v2h xp = {(_Float16)x0, (_Float16)x1};
      *(v2h*)(row + 32) = xp;
    } else {
      float f = 3.14159265358979323846f;
#pragma unroll
      for (int j = 0; j < 10; ++j) {
        float a0 = x0 * f, a1 = x1 * f;
        row[34 + j] = (_Float16)__sinf(a0);
        row[44 + j] = (_Float16)__cosf(a0);
        row[54 + j] = (_Float16)__sinf(a1);
        row[64 + j] = (_Float16)__cosf(a1);
        f *= 2.0f;
      }
      v2h z2 = {(_Float16)0.0f, (_Float16)0.0f};
#pragma unroll
      for (int j = 0; j < 11; ++j) *(v2h*)(row + 74 + 2 * j) = z2;
    }
  }
  lds_fence();

  // ---- 4 layers through WMMA (in-place activation buffer) ----
  layer<3, 8, false>(buf, wf, 0,  b0, lane, nullptr, pbase);
  lds_fence();
  layer<4, 8, false>(buf, wf, 24, b1, lane, nullptr, pbase);
  lds_fence();
  layer<4, 8, false>(buf, wf, 56, b2, lane, nullptr, pbase);
  lds_fence();
  layer<4, 1, true >(buf, wf, 88, b3, lane, out, pbase);
}

// ---------------------------------------------------------------------------
extern "C" void kernel_launch(void* const* d_in, const int* in_sizes, int n_in,
                              void* d_out, int out_size, void* d_ws, size_t ws_size,
                              hipStream_t stream) {
  const float* pos  = (const float*)d_in[0];
  const float* grid = (const float*)d_in[1];
  const float* W0 = (const float*)d_in[2]; const float* b0 = (const float*)d_in[3];
  const float* W1 = (const float*)d_in[4]; const float* b1 = (const float*)d_in[5];
  const float* W2 = (const float*)d_in[6]; const float* b2 = (const float*)d_in[7];
  const float* W3 = (const float*)d_in[8]; const float* b3 = (const float*)d_in[9];
  float* out = (float*)d_out;
  _Float16* wf = (_Float16*)d_ws;   // 92 fragments * 1KB = 94208 bytes

  const int nPoints = in_sizes[0] / 2;
  const int tiles   = (nPoints + TILE_M - 1) / TILE_M;
  const int blocks  = (tiles + WAVES_PER_BLOCK - 1) / WAVES_PER_BLOCK;

  build_wfrags<<<(NUM_FRAGS * 32 + 127) / 128, 128, 0, stream>>>(W0, W1, W2, W3, wf);
  fourier_ngp_mlp<<<blocks, WAVES_PER_BLOCK * 32, 0, stream>>>(
      pos, grid, b0, b1, b2, b3, wf, out, nPoints);
}
